// selfAttentionHead_31799937859875
// MI455X (gfx1250) — compile-verified, matrix operand order checked by reading, not run on hardware
//
#include <hip/hip_runtime.h>
#include <stdint.h>

#define BB 8
#define SS 2048
#define NI 1024
#define ND 128          // NQK == NV == 128
#define QWAVES 4
#define XPITCH 1032     // proj LDS pitch (bf16 elems)

#define KPITCH 136      // k-tile LDS pitch (elems): 272B rows -> 4-bank row skew
#define VPITCH 72       // v-tile LDS pitch (elems): 144B rows
#define PPITCH 72       // P-tile LDS pitch (elems)
#define KS_ELEMS (64 * KPITCH)     // 8704 elems = 17408 B per buffer
#define VS_ELEMS (128 * VPITCH)    // 9216 elems = 18432 B per buffer

typedef __bf16 v16bf __attribute__((ext_vector_type(16)));
typedef float  v8f   __attribute__((ext_vector_type(8)));
typedef uint32_t u32x4 __attribute__((ext_vector_type(4)));

union Frag16 { v16bf v; uint32_t u[8]; u32x4 q[2]; };
union FragC  { v8f v; float f[8]; };

__device__ __forceinline__ uint32_t cvt2(float lo, float hi) {
    uint32_t out;
    asm("v_cvt_pk_bf16_f32 %0, %1, %2" : "=v"(out) : "v"(lo), "v"(hi));
    return out;
}
__device__ __forceinline__ v8f wmma_bf16(const v16bf a, const v16bf b, const v8f c) {
    return __builtin_amdgcn_wmma_f32_16x16x32_bf16(false, a, false, b, (short)0, c,
                                                   false, false);
}
__device__ __forceinline__ void load_frag_bf16(Frag16& f, const uint16_t* p) {
    f.q[0] = *(const u32x4*)(p);
    f.q[1] = *(const u32x4*)(p + 16);
}
__device__ __forceinline__ u32x4 cvt8(const float* p) {
    float4 f0 = *(const float4*)(p);
    float4 f1 = *(const float4*)(p + 4);
    u32x4 pk;
    pk.x = cvt2(f0.x, f0.y); pk.y = cvt2(f0.z, f0.w);
    pk.z = cvt2(f1.x, f1.y); pk.w = cvt2(f1.z, f1.w);
    return pk;
}
// async copy 16B: LDS[lds_byte] <- MEM[sbase + voff]   (ASYNCcnt-tracked)
__device__ __forceinline__ void async_ld16(uint32_t lds_byte, uint32_t voff,
                                           const uint16_t* sbase) {
    asm volatile("global_load_async_to_lds_b128 %0, %1, %2"
                 :: "v"(lds_byte), "v"(voff), "s"(sbase) : "memory");
}

// ---------------------------------------------------------------------------
__global__ __launch_bounds__(256) void cvt_w_kernel(const float* __restrict__ src,
                                                    uint16_t* __restrict__ dst) {
    int i = (blockIdx.x * 256 + threadIdx.x) * 8;
    *(u32x4*)(dst + i) = cvt8(src + i);
}

// ---------------------------------------------------------------------------
// Phase 1: 16x16 tile of oscale*(x @ W^T + b) per wave; bf16 out.
// ---------------------------------------------------------------------------
__global__ __launch_bounds__(256) void qkv_proj_kernel(
    const float* __restrict__ x, const uint16_t* __restrict__ Wbf,
    const float* __restrict__ bias, uint16_t* __restrict__ out,
    float oscale, int store_transposed)
{
    __shared__ uint16_t xs[16 * XPITCH];

    const int tid  = threadIdx.x;
    const int lane = tid & 31;
    const int wv   = tid >> 5;
    const int r    = lane & 15;
    const int half = lane >> 4;
    const int koff = half * 8;
    const int rb   = blockIdx.x;
    const int n0   = wv * 16;

    {   // cooperative stage: x[16][1024] fp32 -> bf16 LDS
        const int srow = tid >> 4;
        const int scol = (tid & 15) * 64;
        const float* xr = x + (size_t)(rb * 16 + srow) * NI + scol;
        uint16_t* xw = xs + srow * XPITCH + scol;
        #pragma unroll
        for (int i = 0; i < 8; ++i)
            *(u32x4*)(xw + i * 8) = cvt8(xr + i * 8);
    }
    __syncthreads();

    const uint16_t* arow = xs + r * XPITCH + koff;
    const uint16_t* wrow = Wbf + (size_t)(n0 + r) * NI + koff;

    FragC c;
    float bvv = bias[n0 + r];
    #pragma unroll
    for (int j = 0; j < 8; ++j) c.f[j] = bvv;

    #pragma unroll 4
    for (int k0 = 0; k0 < NI; k0 += 32) {
        Frag16 a, b;
        load_frag_bf16(a, arow + k0);
        load_frag_bf16(b, wrow + k0);
        c.v = wmma_bf16(a.v, b.v, c.v);
    }

    if (!store_transposed) {
        #pragma unroll
        for (int j = 0; j < 8; j += 2) {
            uint32_t pk = cvt2(c.f[j] * oscale, c.f[j + 1] * oscale);
            int tok = rb * 16 + j + 8 * half;
            out[(size_t)tok * ND + n0 + r]       = (uint16_t)pk;
            out[(size_t)(tok + 1) * ND + n0 + r] = (uint16_t)(pk >> 16);
        }
    } else {
        #pragma unroll
        for (int j = 0; j < 8; j += 2) {
            uint32_t pk = cvt2(c.f[j] * oscale, c.f[j + 1] * oscale);
            int tok = rb * 16 + j + 8 * half;
            int b_  = tok >> 11;
            int s_  = tok & (SS - 1);
            out[((size_t)b_ * ND + (n0 + r)) * SS + s_]     = (uint16_t)pk;
            out[((size_t)b_ * ND + (n0 + r)) * SS + s_ + 1] = (uint16_t)(pk >> 16);
        }
    }
}

// ---------------------------------------------------------------------------
// Phase 2: flash attention, 64-key blocks, async double-buffered k/v in LDS.
// One wave owns 16 query rows x all 128 d. q pre-scaled by log2(e).
// ---------------------------------------------------------------------------
__global__ __launch_bounds__(32 * QWAVES) void attn_kernel(
    const uint16_t* __restrict__ qm,   // [B*S][ND] bf16  (pre-scaled by log2 e)
    const uint16_t* __restrict__ km,   // [B*S][ND] bf16
    const uint16_t* __restrict__ vt,   // [B][ND][S] bf16
    float* __restrict__ y)             // [B][S][ND] f32
{
    __shared__ uint16_t smem[2 * KS_ELEMS + 2 * VS_ELEMS + QWAVES * 16 * PPITCH];

    const int tid  = threadIdx.x;
    const int lane = tid & 31;
    const int wv   = tid >> 5;
    const int r    = lane & 15;
    const int half = lane >> 4;
    const int koff = half * 8;
    const int b_   = blockIdx.y;
    const int qrow0 = (blockIdx.x * QWAVES + wv) * 16;
    const size_t tok0 = (size_t)b_ * SS;

    uint16_t* pw = smem + 2 * KS_ELEMS + 2 * VS_ELEMS + wv * 16 * PPITCH;
    const uint32_t smemB = (uint32_t)(uintptr_t)(&smem[0]);   // LDS byte offset
    const uint32_t kB0 = smemB;                                // k buffers (bytes)
    const uint32_t vB0 = smemB + 2 * KS_ELEMS * 2;             // v buffers (bytes)

    // per-thread async-copy address tables (16B chunk id = tid + 128*i)
    uint32_t klo[8], vlo[8];
    #pragma unroll
    for (int i = 0; i < 8; ++i) {
        int c = tid + 128 * i;
        klo[i] = (uint32_t)((c >> 4) * (KPITCH * 2) + (c & 15) * 16);
        vlo[i] = (uint32_t)((c >> 3) * (VPITCH * 2) + (c & 7) * 16);
    }
    const uint16_t* kgb = km + tok0 * ND;                      // uniform bases
    const uint16_t* vgb = vt + (size_t)b_ * ND * SS;
    const uint32_t kvoff0 = (uint32_t)tid * 16;                // k tile is contiguous
    const uint32_t vvoff0 = (uint32_t)((tid >> 3) * (SS * 2) + (tid & 7) * 16);

    // q A-fragments: 16 rows x 128 d
    Frag16 aq[4];
    {
        const uint16_t* qrow = qm + (tok0 + qrow0 + r) * ND + koff;
        #pragma unroll
        for (int t = 0; t < 4; ++t) load_frag_bf16(aq[t], qrow + t * 32);
    }

    FragC acc[8];
    #pragma unroll
    for (int t = 0; t < 8; ++t)
        #pragma unroll
        for (int j = 0; j < 8; ++j) acc[t].f[j] = 0.f;

    float mstate[8], lsum[8];
    #pragma unroll
    for (int j = 0; j < 8; ++j) { mstate[j] = -3.0e38f; lsum[j] = 0.f; }

    // prime buffer 0 with key block 0
    #pragma unroll
    for (int i = 0; i < 8; ++i) {
        async_ld16(kB0 + klo[i], kvoff0 + i * 2048,  kgb);
        async_ld16(vB0 + vlo[i], vvoff0 + i * 65536, vgb);
    }

    int sel = 0;
    for (int kb = 0; kb < SS; kb += 64) {
        asm volatile("s_wait_asynccnt 0" ::: "memory");
        __syncthreads();                                  // tiles[sel] ready

        if (kb + 64 < SS) {                               // prefetch into tiles[sel^1]
            const uint32_t kvo = kvoff0 + (uint32_t)(kb + 64) * (ND * 2);
            const uint32_t vvo = vvoff0 + (uint32_t)(kb + 64) * 2;
            const uint32_t kBp = kB0 + (sel ^ 1) * (KS_ELEMS * 2);
            const uint32_t vBp = vB0 + (sel ^ 1) * (VS_ELEMS * 2);
            #pragma unroll
            for (int i = 0; i < 8; ++i) {
                async_ld16(kBp + klo[i], kvo + i * 2048,  kgb);
                async_ld16(vBp + vlo[i], vvo + i * 65536, vgb);
            }
        }
        const uint16_t* ksc = smem + sel * KS_ELEMS;
        const uint16_t* vsc = smem + 2 * KS_ELEMS + sel * VS_ELEMS;

        // ---- scores (log2 domain): 4 key-subtiles of 16 x 4 d-windows ----
        FragC s[4];
        #pragma unroll
        for (int u = 0; u < 4; ++u)
            #pragma unroll
            for (int j = 0; j < 8; ++j) s[u].f[j] = 0.f;
        #pragma unroll
        for (int u = 0; u < 4; ++u) {
            const uint16_t* kr = ksc + (u * 16 + r) * KPITCH + koff;
            #pragma unroll
            for (int t = 0; t < 4; ++t) {
                Frag16 bk;
                load_frag_bf16(bk, kr + t * 32);
                s[u].v = wmma_bf16(aq[t].v, bk.v, s[u].v);
            }
        }

        // ---- online softmax, P -> LDS ----
        float fscale[8];
        #pragma unroll
        for (int j = 0; j < 8; ++j) {
            float m = fmaxf(fmaxf(s[0].f[j], s[1].f[j]), fmaxf(s[2].f[j], s[3].f[j]));
            m = fmaxf(m, __shfl_xor(m, 1, 32));
            m = fmaxf(m, __shfl_xor(m, 2, 32));
            m = fmaxf(m, __shfl_xor(m, 4, 32));
            m = fmaxf(m, __shfl_xor(m, 8, 32));
            float mnew = fmaxf(mstate[j], m);
            float fs = __builtin_exp2f(mstate[j] - mnew);
            float p0 = __builtin_exp2f(s[0].f[j] - mnew);
            float p1 = __builtin_exp2f(s[1].f[j] - mnew);
            float p2 = __builtin_exp2f(s[2].f[j] - mnew);
            float p3 = __builtin_exp2f(s[3].f[j] - mnew);
            float ps = (p0 + p1) + (p2 + p3);
            ps += __shfl_xor(ps, 1, 32);
            ps += __shfl_xor(ps, 2, 32);
            ps += __shfl_xor(ps, 4, 32);
            ps += __shfl_xor(ps, 8, 32);
            lsum[j] = lsum[j] * fs + ps;
            mstate[j] = mnew;
            fscale[j] = fs;
            int rr = j + 8 * half;
            uint32_t pa = cvt2(p0, p1);
            uint32_t pb = cvt2(p2, p3);
            pw[rr * PPITCH + r]      = (uint16_t)pa;
            pw[rr * PPITCH + 16 + r] = (uint16_t)(pa >> 16);
            pw[rr * PPITCH + 32 + r] = (uint16_t)pb;
            pw[rr * PPITCH + 48 + r] = (uint16_t)(pb >> 16);
        }

        #pragma unroll
        for (int t = 0; t < 8; ++t)
            #pragma unroll
            for (int j = 0; j < 8; ++j) acc[t].f[j] *= fscale[j];

        asm volatile("s_wait_dscnt 0" ::: "memory");      // P visible to own wave

        Frag16 ap[2];
        #pragma unroll
        for (int c = 0; c < 2; ++c) {
            ap[c].q[0] = *(const u32x4*)(pw + r * PPITCH + 32 * c + koff);
            ap[c].q[1] = *(const u32x4*)(pw + r * PPITCH + 32 * c + koff + 16);
        }

        // ---- y += P @ V ----
        #pragma unroll
        for (int t = 0; t < 8; ++t) {
            const uint16_t* vr = vsc + (t * 16 + r) * VPITCH;
            #pragma unroll
            for (int c = 0; c < 2; ++c) {
                Frag16 bv;
                load_frag_bf16(bv, vr + 32 * c + koff);
                acc[t].v = wmma_bf16(ap[c].v, bv.v, acc[t].v);
            }
        }
        asm volatile("" ::: "memory");
        sel ^= 1;
    }

    float inv[8];
    #pragma unroll
    for (int j = 0; j < 8; ++j) inv[j] = 1.0f / lsum[j];
    #pragma unroll
    for (int t = 0; t < 8; ++t)
        #pragma unroll
        for (int j = 0; j < 8; ++j) {
            int row = qrow0 + j + 8 * half;
            y[(tok0 + row) * ND + t * 16 + r] = acc[t].f[j] * inv[j];
        }
}

// ---------------------------------------------------------------------------
extern "C" void kernel_launch(void* const* d_in, const int* in_sizes, int n_in,
                              void* d_out, int out_size, void* d_ws, size_t ws_size,
                              hipStream_t stream) {
    const float* x  = (const float*)d_in[0];
    const float* Wq = (const float*)d_in[1];
    const float* bq = (const float*)d_in[2];
    const float* Wk = (const float*)d_in[3];
    const float* bk = (const float*)d_in[4];
    const float* Wv = (const float*)d_in[5];
    const float* bv = (const float*)d_in[6];
    float* y = (float*)d_out;

    uint16_t* qm  = (uint16_t*)d_ws;
    uint16_t* km  = qm  + (size_t)BB * SS * ND;
    uint16_t* vt  = km  + (size_t)BB * SS * ND;
    uint16_t* wqb = vt  + (size_t)BB * SS * ND;
    uint16_t* wkb = wqb + (size_t)ND * NI;
    uint16_t* wvb = wkb + (size_t)ND * NI;

    const float LOG2E = 1.4426950408889634f;

    dim3 wgrid(ND * NI / (256 * 8));
    cvt_w_kernel<<<wgrid, 256, 0, stream>>>(Wq, wqb);
    cvt_w_kernel<<<wgrid, 256, 0, stream>>>(Wk, wkb);
    cvt_w_kernel<<<wgrid, 256, 0, stream>>>(Wv, wvb);

    dim3 pgrid(BB * SS / 16);
    qkv_proj_kernel<<<pgrid, 256, 0, stream>>>(x, wqb, bq, qm, LOG2E, 0);
    qkv_proj_kernel<<<pgrid, 256, 0, stream>>>(x, wkb, bk, km, 1.0f, 0);
    qkv_proj_kernel<<<pgrid, 256, 0, stream>>>(x, wvb, bv, vt, 1.0f, 1);

    dim3 agrid(SS / (16 * QWAVES), BB);
    attn_kernel<<<agrid, 32 * QWAVES, 0, stream>>>(qm, km, vt, y);
}
